// GPS_26792005992742
// MI455X (gfx1250) — matrix operand order repaired and checked by reading, not compiled, not run on hardware
//
#include <hip/hip_runtime.h>
#include <hip/hip_bf16.h>
#include <math.h>

// ---------------------------------------------------------------------------
// GraphGPS forward (3 layers) for MI455X / gfx1250, wave32 + WMMA f16->f32.
// v3: softmax without per-thread spill array; fused BN(h)+BN(ha) add kernel.
// ---------------------------------------------------------------------------

typedef __attribute__((ext_vector_type(16))) _Float16 v16h;
typedef __attribute__((ext_vector_type(8)))  float    v8f;

#define N_NODES 65536
#define SEQ     128
#define CH      128
#define N_EDGES 262144
#define PE_IN   20
#define PE_OUT  8
#define EMB_DIM 120
#define EPS_BN  1e-5f
#define APAD    34      // 32-wide K tile padded to 34 halves (bank-conflict free)

// ----------------------------- helpers -------------------------------------

__device__ inline unsigned pack2h(float x, float y) {
  union { unsigned u; _Float16 h[2]; } p;
  p.h[0] = (_Float16)x; p.h[1] = (_Float16)y;
  return p.u;
}

__device__ inline v8f wmma_f16(v16h a, v16h b, v8f c) {
  // (neg_a, A, neg_b, B, c_mod, C, reuse_a, reuse_b)
  return __builtin_amdgcn_wmma_f32_16x16x32_f16(false, a, false, b, (short)0, c,
                                                false, false);
}

// A-fragment (16x32 f16, M x K) from LDS row-major [row][k] with `stride` halves.
// Lane L<16: M=L, K=0..15; lane L>=16: M=L-16, K=16..31.
__device__ inline v16h load_afrag(const _Float16* As, int rt, int lane, int stride) {
  union { v16h v; unsigned u[8]; } f;
  const int m  = rt * 16 + (lane & 15);
  const int kb = (lane < 16) ? 0 : 16;
  const _Float16* row = As + m * stride;
#pragma unroll
  for (int j = 0; j < 8; ++j) {
    int k = kb + (j << 1);
    f.u[j] = *(const unsigned*)(row + k);
  }
  return f.v;
}

// B-fragment (32x16 f16, K x N) from memory stored TRANSPOSED as [n][k].
// Lane L<16: N=L, K=0..15; lane L>=16: N=L-16, K=16..31.
__device__ inline v16h load_bfrag_lds(const _Float16* Bs, int ct, int lane, int stride) {
  union { v16h v; unsigned u[8]; } f;
  const int n  = ct * 16 + (lane & 15);
  const int kb = (lane < 16) ? 0 : 16;
  const _Float16* col = Bs + n * stride + kb;
#pragma unroll
  for (int j = 0; j < 8; ++j) f.u[j] = *(const unsigned*)(col + 2 * j);
  return f.v;
}

// ----------------------------- weight convert ------------------------------
// wt[n*K + k] = (f16) w[k*Nout + n]   (one-time, tiny)
__global__ __launch_bounds__(256) void convw_kernel(
    const float* __restrict__ w, _Float16* __restrict__ wt, int K, int Nout) {
  int i = blockIdx.x * blockDim.x + threadIdx.x;
  if (i >= K * Nout) return;
  int n = i / K, k = i - n * K;
  wt[i] = (_Float16)w[(unsigned)k * Nout + n];
}

// ----------------------------- BN stats ------------------------------------
__global__ __launch_bounds__(256) void bn_stats_kernel(
    const float* __restrict__ in, int C, int rowsPerBlock,
    float* __restrict__ gsum, float* __restrict__ gsq) {
  __shared__ float ssum[384];
  __shared__ float ssq[384];
  const int t = threadIdx.x;
  if (t < C) { ssum[t] = 0.f; ssq[t] = 0.f; }
  __syncthreads();
  const int r0  = blockIdx.x * rowsPerBlock;
  const int nth = blockDim.x / C;
  const int c   = t % C;
  const int tr  = t / C;
  if (tr < nth) {
    float s = 0.f, q = 0.f;
    for (int r = r0 + tr; r < r0 + rowsPerBlock; r += nth) {
      float v = in[(unsigned)r * C + c];
      s += v; q += v * v;
    }
    atomicAdd(&ssum[c], s);
    atomicAdd(&ssq[c], q);
  }
  __syncthreads();
  if (t < C) {
    atomicAdd(&gsum[t], ssum[t]);
    atomicAdd(&gsq[t],  ssq[t]);
  }
}

__global__ __launch_bounds__(256) void bn_apply_kernel(
    const float* __restrict__ in, float* __restrict__ out,
    const float* __restrict__ g, const float* __restrict__ b,
    const float* __restrict__ gsum, const float* __restrict__ gsq,
    int C, int total, float invN) {
  int i = blockIdx.x * blockDim.x + threadIdx.x;
  if (i >= total) return;
  int c = i % C;
  float mean = gsum[c] * invN;
  float var  = gsq[c] * invN - mean * mean;
  out[i] = (in[i] - mean) * rsqrtf(var + EPS_BN) * g[c] + b[c];
}

// out = BN_a(a) + BN_c(c)  (fused local + attention branch combine)
__global__ __launch_bounds__(256) void bn2_add_kernel(
    const float* __restrict__ a, const float* __restrict__ c,
    const float* __restrict__ ga, const float* __restrict__ ba,
    const float* __restrict__ gc, const float* __restrict__ bc,
    const float* __restrict__ sA, const float* __restrict__ qA,
    const float* __restrict__ sC, const float* __restrict__ qC,
    float* __restrict__ out, int C, int total, float invN) {
  int i = blockIdx.x * blockDim.x + threadIdx.x;
  if (i >= total) return;
  int ch = i % C;
  float mA = sA[ch] * invN, vA = qA[ch] * invN - mA * mA;
  float mC = sC[ch] * invN, vC = qC[ch] * invN - mC * mC;
  float h  = (a[i] - mA) * rsqrtf(vA + EPS_BN) * ga[ch] + ba[ch];
  float ha = (c[i] - mC) * rsqrtf(vC + EPS_BN) * gc[ch] + bc[ch];
  out[i] = h + ha;
}

// ----------------------------- input features ------------------------------
__global__ __launch_bounds__(256) void build_x0_kernel(
    const int* __restrict__ node_type, const float* __restrict__ pe,
    const float* __restrict__ gamma, const float* __restrict__ beta,
    const float* __restrict__ peW, const float* __restrict__ peB,
    const float* __restrict__ node_emb,
    const float* __restrict__ gsum, const float* __restrict__ gsq,
    float* __restrict__ x) {
  int i = blockIdx.x * blockDim.x + threadIdx.x;
  int n = i >> 7, c = i & 127;
  const float invN = 1.0f / (float)N_NODES;
  float v;
  if (c < EMB_DIM) {
    v = node_emb[(unsigned)node_type[n] * EMB_DIM + c];
  } else {
    int j = c - EMB_DIM;
    float acc = peB[j];
#pragma unroll
    for (int k = 0; k < PE_IN; ++k) {
      float mean = gsum[k] * invN;
      float var  = gsq[k] * invN - mean * mean;
      float p = (pe[(unsigned)n * PE_IN + k] - mean) * rsqrtf(var + EPS_BN) * gamma[k] + beta[k];
      acc += p * peW[k * PE_OUT + j];
    }
    v = acc;
  }
  x[(unsigned)n * CH + c] = v;
}

// ----------------------------- GINE scatter --------------------------------
__global__ __launch_bounds__(256) void scatter_kernel(
    const float* __restrict__ x, const int* __restrict__ ei,
    const int* __restrict__ et, const float* __restrict__ edge_emb,
    float* __restrict__ agg) {
  int gid = blockIdx.x * blockDim.x + threadIdx.x;
  int e = gid >> 5, g = gid & 31;
  if (e >= N_EDGES) return;
  int src = ei[e];
  int dst = ei[N_EDGES + e];
  int ty  = et[e];
  const float4 xv = *(const float4*)&x[(unsigned)src * CH + g * 4];
  const float4 ev = *(const float4*)&edge_emb[(unsigned)ty * CH + g * 4];
  float4 m;
  m.x = fmaxf(xv.x + ev.x, 0.f);
  m.y = fmaxf(xv.y + ev.y, 0.f);
  m.z = fmaxf(xv.z + ev.z, 0.f);
  m.w = fmaxf(xv.w + ev.w, 0.f);
  float* dp = &agg[(unsigned)dst * CH + g * 4];
  atomicAdd(dp + 0, m.x);
  atomicAdd(dp + 1, m.y);
  atomicAdd(dp + 2, m.z);
  atomicAdd(dp + 3, m.w);
}

// ----------------------------- generic WMMA GEMM ---------------------------
// Y[M, Nout] = act( (A (+ Aadd)) @ W + bias (+ Radd) ), W given pre-transposed
// f16 as Wt[Nout][K]. 256 threads = 8 waves; block tile 128x128.
// A tiles staged f32->f16 into LDS with a 1-deep software pipeline; B frags
// loaded straight from (L2-resident) Wt.
__global__ __launch_bounds__(256) void gemm_kernel(
    const float* __restrict__ A, const float* __restrict__ Aadd,
    const _Float16* __restrict__ Wt, const float* __restrict__ bias,
    const float* __restrict__ Radd, float* __restrict__ Y,
    int K, int Nout, int relu) {
  __shared__ _Float16 As[128 * APAD];
  const int t = threadIdx.x, wave = t >> 5, lane = t & 31;
  const int row0 = blockIdx.x * 128;
  const int col0 = blockIdx.y * 128;
  const bool hasAdd = (Aadd != nullptr);

  v8f acc[8];
#pragma unroll
  for (int ct = 0; ct < 8; ++ct) acc[ct] = (v8f){0.f,0.f,0.f,0.f,0.f,0.f,0.f,0.f};

  float4 pa[4], pb[4];
  // prologue: load first A tile into registers
#pragma unroll
  for (int it = 0; it < 4; ++it) {
    int idx = it * 256 + t;
    int r = idx >> 3, c4 = (idx & 7) << 2;
    unsigned off = (unsigned)(row0 + r) * K + c4;
    pa[it] = *(const float4*)(A + off);
    if (hasAdd) pb[it] = *(const float4*)(Aadd + off);
  }

  const int ksteps = K >> 5;
  for (int ks = 0; ks < ksteps; ++ks) {
    // commit staged registers to LDS (pack f32 pairs -> b32 stores)
#pragma unroll
    for (int it = 0; it < 4; ++it) {
      int idx = it * 256 + t;
      int r = idx >> 3, c4 = (idx & 7) << 2;
      float4 v = pa[it];
      if (hasAdd) { v.x += pb[it].x; v.y += pb[it].y; v.z += pb[it].z; v.w += pb[it].w; }
      *(unsigned*)(As + r * APAD + c4)     = pack2h(v.x, v.y);
      *(unsigned*)(As + r * APAD + c4 + 2) = pack2h(v.z, v.w);
    }
    __syncthreads();
    // issue next tile's global loads; they overlap the WMMAs below
    if (ks + 1 < ksteps) {
      int k0n = (ks + 1) << 5;
#pragma unroll
      for (int it = 0; it < 4; ++it) {
        int idx = it * 256 + t;
        int r = idx >> 3, c4 = (idx & 7) << 2;
        unsigned off = (unsigned)(row0 + r) * K + k0n + c4;
        pa[it] = *(const float4*)(A + off);
        if (hasAdd) pb[it] = *(const float4*)(Aadd + off);
      }
    }
    v16h a = load_afrag(As, wave, lane, APAD);
    const int kbB = ((lane < 16) ? 0 : 16) + (ks << 5);
#pragma unroll
    for (int ct = 0; ct < 8; ++ct) {
      union { v16h v; unsigned u[8]; } bf;
      const _Float16* wp = Wt + (unsigned)(col0 + ct * 16 + (lane & 15)) * K + kbB;
#pragma unroll
      for (int j = 0; j < 8; ++j) bf.u[j] = *(const unsigned*)(wp + 2 * j);
      acc[ct] = wmma_f16(a, bf.v, acc[ct]);
    }
    __syncthreads();
  }

  const int n0 = lane & 15;
#pragma unroll
  for (int ct = 0; ct < 8; ++ct) {
    int col = col0 + ct * 16 + n0;
    float bv = bias ? bias[col] : 0.f;
#pragma unroll
    for (int r = 0; r < 8; ++r) {
      int row = row0 + wave * 16 + r + ((lane < 16) ? 0 : 8);
      float v = acc[ct][r] + bv;
      if (Radd) v += Radd[(unsigned)row * Nout + col];
      if (relu) v = fmaxf(v, 0.f);
      Y[(unsigned)row * Nout + col] = v;
    }
  }
}

// ----------------------------- attention -----------------------------------
// grid: (512 graphs, 4 = head*2 + rowhalf). Block handles 64 query rows of one
// head of one graph. qkv layout [N, 384] = [q | k | v].
// LDS overlay (58368 B total):
//   region X (33024 B): Sf 64x129 f32        -> later Vt 64x130 f16
//   region Y (25344 B): Qs 64x66 + Ks 128x66 -> later Pm 64x130 f16
#define SFP  129
#define VPP  130
#define QKP  66
__global__ __launch_bounds__(256) void attn_kernel(
    const float* __restrict__ qkv, float* __restrict__ o) {
  __shared__ char smem[33024 + 25344];
  float*    Sf = (float*)smem;
  _Float16* Vt = (_Float16*)smem;
  _Float16* Qs = (_Float16*)(smem + 33024);
  _Float16* Ks = (_Float16*)(smem + 33024 + 64 * QKP * 2);
  _Float16* Pm = (_Float16*)(smem + 33024);

  const int g = blockIdx.x;
  const int head = blockIdx.y >> 1;
  const int half = blockIdx.y & 1;
  const int t = threadIdx.x, wave = t >> 5, lane = t & 31;
  const int gbase = g * SEQ;
  const int rowbase = gbase + half * 64;
  const float scale = 0.125f;  // 1/sqrt(64)

  // --- stage Q (64x64) and K (128x64) as f16 ---
#pragma unroll
  for (int it = 0; it < 4; ++it) {   // Q: 1024 float4
    int idx = it * 256 + t;
    int r = idx >> 4, d4 = (idx & 15) << 2;
    float4 f = *(const float4*)(qkv + (unsigned)(rowbase + r) * 384 + head * 64 + d4);
    *(unsigned*)(Qs + r * QKP + d4)     = pack2h(f.x, f.y);
    *(unsigned*)(Qs + r * QKP + d4 + 2) = pack2h(f.z, f.w);
  }
#pragma unroll
  for (int it = 0; it < 8; ++it) {   // K: 2048 float4
    int idx = it * 256 + t;
    int r = idx >> 4, d4 = (idx & 15) << 2;
    float4 f = *(const float4*)(qkv + (unsigned)(gbase + r) * 384 + 128 + head * 64 + d4);
    *(unsigned*)(Ks + r * QKP + d4)     = pack2h(f.x, f.y);
    *(unsigned*)(Ks + r * QKP + d4 + 2) = pack2h(f.z, f.w);
  }
  __syncthreads();

  // --- Phase 1: S = (Q K^T) * scale -- 4x8 tiles of 16x16, 4 per wave ---
#pragma unroll
  for (int i = 0; i < 4; ++i) {
    int tid = wave * 4 + i, rt = tid >> 3, ct = tid & 7;
    v8f acc = (v8f){0.f,0.f,0.f,0.f,0.f,0.f,0.f,0.f};
#pragma unroll
    for (int ks = 0; ks < 2; ++ks) {
      v16h a = load_afrag(Qs + ks * 32, rt, lane, QKP);
      v16h b = load_bfrag_lds(Ks + ks * 32, ct, lane, QKP);
      acc = wmma_f16(a, b, acc);
    }
    int n = lane & 15;
#pragma unroll
    for (int r = 0; r < 8; ++r) {
      int m = rt * 16 + r + ((lane < 16) ? 0 : 8);
      Sf[m * SFP + ct * 16 + n] = acc[r] * scale;
    }
  }
  __syncthreads();

  // --- Phase 2: row softmax (each row owned by one thread; exp in place) ---
  if (t < 64) {
    float* row = Sf + t * SFP;
    float mx = -1e30f;
    for (int c = 0; c < SEQ; ++c) mx = fmaxf(mx, row[c]);
    float s = 0.f;
    for (int c = 0; c < SEQ; ++c) { float e = __expf(row[c] - mx); row[c] = e; s += e; }
    float inv = 1.0f / s;
    _Float16* prow = Pm + t * VPP;
    for (int c = 0; c < SEQ; c += 2)
      *(unsigned*)(prow + c) = pack2h(row[c] * inv, row[c + 1] * inv);
  }
  __syncthreads();

  // --- stage V transposed [dim][seq] over the Sf region ---
#pragma unroll
  for (int it = 0; it < 8; ++it) {   // 2048 float4
    int idx = it * 256 + t;
    int s = idx >> 4, d4 = (idx & 15) << 2;
    float4 f = *(const float4*)(qkv + (unsigned)(gbase + s) * 384 + 256 + head * 64 + d4);
    Vt[(d4 + 0) * VPP + s] = (_Float16)f.x;
    Vt[(d4 + 1) * VPP + s] = (_Float16)f.y;
    Vt[(d4 + 2) * VPP + s] = (_Float16)f.z;
    Vt[(d4 + 3) * VPP + s] = (_Float16)f.w;
  }
  __syncthreads();

  // --- Phase 3: O = P @ V -- 4x4 tiles of 16x16, 2 per wave, K(seq)=128 ---
#pragma unroll
  for (int i = 0; i < 2; ++i) {
    int tid = wave * 2 + i, rt = tid >> 2, cv = tid & 3;
    v8f acc = (v8f){0.f,0.f,0.f,0.f,0.f,0.f,0.f,0.f};
#pragma unroll
    for (int ks = 0; ks < 4; ++ks) {
      v16h a = load_afrag(Pm + ks * 32, rt, lane, VPP);
      v16h b = load_bfrag_lds(Vt + ks * 32, cv, lane, VPP);
      acc = wmma_f16(a, b, acc);
    }
    int n = lane & 15;
#pragma unroll
    for (int r = 0; r < 8; ++r) {
      int m = rt * 16 + r + ((lane < 16) ? 0 : 8);
      o[(unsigned)(rowbase + m) * CH + head * 64 + cv * 16 + n] = acc[r];
    }
  }
}

// ----------------------------- launcher -------------------------------------
extern "C" void kernel_launch(void* const* d_in, const int* in_sizes, int n_in,
                              void* d_out, int out_size, void* d_ws, size_t ws_size,
                              hipStream_t stream) {
  (void)in_sizes; (void)n_in; (void)out_size; (void)ws_size;

  const int*   node_type = (const int*)  d_in[0];
  const float* pe        = (const float*)d_in[1];
  const int*   edge_idx  = (const int*)  d_in[2];
  const int*   edge_ty   = (const int*)  d_in[3];
  // d_in[4] = batch (implicit in equal-size reshape; unused)
  const float* pe_gamma  = (const float*)d_in[5];
  const float* pe_beta   = (const float*)d_in[6];
  const float* pe_W      = (const float*)d_in[7];
  const float* pe_b      = (const float*)d_in[8];
  const float* node_emb  = (const float*)d_in[9];
  const float* edge_emb  = (const float*)d_in[10];
  const float* gine_W1   = (const float*)d_in[11];
  const float* gine_b1   = (const float*)d_in[12];
  const float* gine_W2   = (const float*)d_in[13];
  const float* gine_b2   = (const float*)d_in[14];
  const float* n1_g      = (const float*)d_in[15];
  const float* n1_b      = (const float*)d_in[16];
  const float* n2_g      = (const float*)d_in[17];
  const float* n2_b      = (const float*)d_in[18];
  const float* n3_g      = (const float*)d_in[19];
  const float* n3_b      = (const float*)d_in[20];
  const float* attn_inW  = (const float*)d_in[21];
  const float* attn_inb  = (const float*)d_in[22];
  const float* attn_outW = (const float*)d_in[23];
  const float* attn_outb = (const float*)d_in[24];
  const float* mlp_W1    = (const float*)d_in[25];
  const float* mlp_b1    = (const float*)d_in[26];
  const float* mlp_W2    = (const float*)d_in[27];
  const float* mlp_b2    = (const float*)d_in[28];

  float* x    = (float*)d_out;                       // current features [N,128]
  float* bufA = (float*)d_ws;                        // [N,128] agg / local branch
  float* bufB = bufA + (size_t)N_NODES * CH;         // [N,128]
  float* bufC = bufB + (size_t)N_NODES * CH;         // [N,128]
  float* bufQ = bufC + (size_t)N_NODES * CH;         // [N,384] qkv / mlp hidden
  float* stA  = bufQ + (size_t)N_NODES * 384;        // sumA[128] | sqA[128]
  float* stC  = stA + 256;                           // sumC[128] | sqC[128]
  _Float16* wt = (_Float16*)(stC + 256);             // f16 transposed weights

  // per-layer f16 weight slabs (halves): [Nout][K] layouts
  const size_t WT_L = 163840;                        // halves per layer
  const size_t O_GW1 = 0, O_GW2 = 16384, O_INW = 32768, O_OUTW = 81920,
               O_MW1 = 98304, O_MW2 = 131072;

  const size_t xbytes = (size_t)N_NODES * CH * sizeof(float);
  const float invN = 1.0f / (float)N_NODES;
  const int totNC = N_NODES * CH;
  dim3 blk(256);
  dim3 grid_ew(totNC / 256);
  dim3 grid_gemm1(N_NODES / 128, 1);
  dim3 grid_gemm2(N_NODES / 128, 2);
  dim3 grid_gemm3(N_NODES / 128, 3);

  // ---- one-time: convert + transpose all weights to f16 [Nout][K] ----
  for (int l = 0; l < 3; ++l) {
    _Float16* w = wt + (size_t)l * WT_L;
    convw_kernel<<<64,  blk, 0, stream>>>(gine_W1   + (size_t)l*CH*CH,  w + O_GW1,  CH,  CH);
    convw_kernel<<<64,  blk, 0, stream>>>(gine_W2   + (size_t)l*CH*CH,  w + O_GW2,  CH,  CH);
    convw_kernel<<<192, blk, 0, stream>>>(attn_inW  + (size_t)l*CH*384, w + O_INW,  CH,  384);
    convw_kernel<<<64,  blk, 0, stream>>>(attn_outW + (size_t)l*CH*CH,  w + O_OUTW, CH,  CH);
    convw_kernel<<<128, blk, 0, stream>>>(mlp_W1    + (size_t)l*CH*256, w + O_MW1,  CH,  256);
    convw_kernel<<<128, blk, 0, stream>>>(mlp_W2    + (size_t)l*256*CH, w + O_MW2,  256, CH);
  }

  // ---- pe BatchNorm stats + input features ----
  hipMemsetAsync(stA, 0, 256 * sizeof(float), stream);
  bn_stats_kernel<<<256, blk, 0, stream>>>(pe, PE_IN, N_NODES / 256, stA, stA + 128);
  build_x0_kernel<<<grid_ew, blk, 0, stream>>>(node_type, pe, pe_gamma, pe_beta,
                                               pe_W, pe_b, node_emb, stA, stA + 128, x);

  for (int l = 0; l < 3; ++l) {
    _Float16* w = wt + (size_t)l * WT_L;
    const float* gb1 = gine_b1 + l * CH;
    const float* gb2 = gine_b2 + l * CH;
    const float* ib  = attn_inb  + l * 384;
    const float* ob  = attn_outb + l * CH;
    const float* mb1 = mlp_b1 + l * 256;
    const float* mb2 = mlp_b2 + l * CH;

    // ---- GINE local branch (pre-BN result in bufA) ----
    hipMemsetAsync(bufA, 0, xbytes, stream);
    scatter_kernel<<<(N_EDGES * 32) / 256, blk, 0, stream>>>(x, edge_idx, edge_ty,
                                                             edge_emb, bufA);
    // bufB = relu((x + agg) @ W1 + b1)
    gemm_kernel<<<grid_gemm1, blk, 0, stream>>>(x, bufA, w + O_GW1, gb1, nullptr,
                                                bufB, CH, CH, 1);
    // bufA = bufB @ W2 + b2 + x
    gemm_kernel<<<grid_gemm1, blk, 0, stream>>>(bufB, nullptr, w + O_GW2, gb2, x,
                                                bufA, CH, CH, 0);
    hipMemsetAsync(stA, 0, 256 * sizeof(float), stream);
    bn_stats_kernel<<<256, blk, 0, stream>>>(bufA, CH, 256, stA, stA + 128);

    // ---- attention branch (pre-BN result in bufC) ----
    gemm_kernel<<<grid_gemm3, blk, 0, stream>>>(x, nullptr, w + O_INW, ib, nullptr,
                                                bufQ, CH, 384, 0);
    attn_kernel<<<dim3(N_NODES / SEQ, 4), blk, 0, stream>>>(bufQ, bufB);
    gemm_kernel<<<grid_gemm1, blk, 0, stream>>>(bufB, nullptr, w + O_OUTW, ob, x,
                                                bufC, CH, CH, 0);
    hipMemsetAsync(stC, 0, 256 * sizeof(float), stream);
    bn_stats_kernel<<<256, blk, 0, stream>>>(bufC, CH, 256, stC, stC + 128);

    // ---- fused combine: bufB = BN1(bufA) + BN2(bufC) ----
    bn2_add_kernel<<<grid_ew, blk, 0, stream>>>(bufA, bufC,
                                                n1_g + l * CH, n1_b + l * CH,
                                                n2_g + l * CH, n2_b + l * CH,
                                                stA, stA + 128, stC, stC + 128,
                                                bufB, CH, totNC, invN);

    // ---- FFN + final BN -> x ----
    gemm_kernel<<<grid_gemm2, blk, 0, stream>>>(bufB, nullptr, w + O_MW1, mb1,
                                                nullptr, bufQ, CH, 256, 1);
    gemm_kernel<<<grid_gemm1, blk, 0, stream>>>(bufQ, nullptr, w + O_MW2, mb2,
                                                bufB, bufC, 256, CH, 0);
    hipMemsetAsync(stA, 0, 256 * sizeof(float), stream);
    bn_stats_kernel<<<256, blk, 0, stream>>>(bufC, CH, 256, stA, stA + 128);
    bn_apply_kernel<<<grid_ew, blk, 0, stream>>>(bufC, x, n3_g + l * CH,
                                                 n3_b + l * CH, stA, stA + 128, CH,
                                                 totNC, invN);
  }
}
